// Block_67989332296091
// MI455X (gfx1250) — matrix operand order repaired and checked by reading, not compiled, not run on hardware
//
#include <hip/hip_runtime.h>
#include <math.h>

#define N_NODES 50000
#define N_EDGES 500000
#define DIN 128
#define DOUT 128

typedef __attribute__((ext_vector_type(16))) __bf16 v16bf;
typedef __attribute__((ext_vector_type(8)))  float  v8f;
typedef __attribute__((ext_vector_type(4)))  int    v4i;

// ------------------------------------------------------------------ async LDS
#if defined(__has_builtin)
#if __has_builtin(__builtin_amdgcn_global_load_async_to_lds_b128) && \
    __has_builtin(__builtin_amdgcn_s_wait_asynccnt)
#define USE_ASYNC_LDS 1
#endif
#endif

// copy 16 contiguous bytes global -> LDS (async on gfx1250, sync fallback)
__device__ __forceinline__ void cp_b128(const void* g, void* l) {
#ifdef USE_ASYNC_LDS
    __builtin_amdgcn_global_load_async_to_lds_b128((v4i*)g, (v4i*)l, 0, 0);
#else
    *(uint4*)l = *(const uint4*)g;
#endif
}
__device__ __forceinline__ void cp_wait() {
#ifdef USE_ASYNC_LDS
    __builtin_amdgcn_s_wait_asynccnt(0);
#endif
}

__device__ __forceinline__ int imin(int a, int b) { return a < b ? a : b; }

// K offset inside a 32-wide chunk for A-fragment element i, lane-half h
// (ISA 05_wmma: 16-bit A 16x32: lanes0-15 VGPR0..3 = K0..7, VGPR4..7 = K16..23;
//  lanes16-31 VGPR0..3 = K8..15, VGPR4..7 = K24..31)
__device__ __forceinline__ int a_koff(int i, int half) {
    return ((i >> 3) << 4) + half * 8 + (i & 7);
}

// A fragment from a row-major f32 row (convert to bf16 on the fly)
__device__ __forceinline__ v16bf load_a_frag_f32(const float* __restrict__ row,
                                                 int kb, int half) {
    v16bf a;
    const float* p0 = row + kb * 32 + half * 8;        // K = kb*32 + h*8 + 0..7
    const float* p1 = row + kb * 32 + 16 + half * 8;   // K = kb*32 + 16 + h*8 + 0..7
#pragma unroll
    for (int i = 0; i < 8; ++i) a[i] = (__bf16)p0[i];
#pragma unroll
    for (int i = 0; i < 8; ++i) a[8 + i] = (__bf16)p1[i];
    return a;
}

// A fragment from a row-major bf16 LDS row
__device__ __forceinline__ v16bf load_a_frag_lds(const __bf16* row, int kb, int half) {
    v16bf a;
    const __bf16* p0 = row + kb * 32 + half * 8;
    const __bf16* p1 = row + kb * 32 + 16 + half * 8;
#pragma unroll
    for (int i = 0; i < 8; ++i) a[i] = p0[i];
#pragma unroll
    for (int i = 0; i < 8; ++i) a[8 + i] = p1[i];
    return a;
}

// B fragment from LDS weights stored transposed [n][k] (ldk = padded K).
// Lane n = L%16; elems 0..7 K = base+h*8+0..7, elems 8..15 K = base+16+h*8+0..7.
// Two contiguous 16B runs.
__device__ __forceinline__ v16bf load_b_frag(const __bf16* sWt, int ldk,
                                             int nt, int kb, int lane) {
    int n = lane & 15, half = lane >> 4;
    const __bf16* p = sWt + (nt * 16 + n) * ldk + kb * 32 + half * 8;
    v16bf b;
#pragma unroll
    for (int i = 0; i < 8; ++i) { b[i] = p[i]; b[8 + i] = p[16 + i]; }
    return b;
}

__device__ __forceinline__ v8f wmma_bf16(v16bf a, v16bf b, v8f c) {
    return __builtin_amdgcn_wmma_f32_16x16x32_bf16(false, a, false, b,
                                                   (short)0, c, false, false);
}

// ---------------------------------------------------------------- zero agg
__global__ void k_zero(float* __restrict__ p, int n) {
    int i = blockIdx.x * blockDim.x + threadIdx.x;
    if (i < n) p[i] = 0.f;
}

// ------------------------------------------------- delta = mlp_h(x)  [N,3]
__global__ void __launch_bounds__(256) k_node_h(
    const float* __restrict__ x,  const float* __restrict__ h_w1,
    const float* __restrict__ h_b1, const float* __restrict__ h_w2,
    const float* __restrict__ h_b2, float* __restrict__ delta)
{
    __shared__ __align__(32) __bf16 sW1t[DIN * DIN];   // transposed [n][k], 32KB
    __shared__ float sW2[DIN * 3];
    __shared__ float sB1[DIN];
    __shared__ float sB2[4];
    __shared__ __align__(32) union {
        float  stage[DIN * DIN];     // 64KB raw f32 weight staging (async copy)
        __bf16 H[8][16 * DIN];       // per-wave hidden tile (after staging done)
    } u;

    int tid = threadIdx.x;
    // contiguous async copy of h_w1 into LDS staging
    for (int c = tid; c < (DIN * DIN) / 4; c += 256)
        cp_b128(h_w1 + c * 4, u.stage + c * 4);
    cp_wait();
    __syncthreads();
    // transpose + convert LDS->LDS (strided reads are cheap in LDS)
    for (int idx = tid; idx < DIN * DIN; idx += 256) {
        int n = idx & 127, k = idx >> 7;
        sW1t[n * DIN + k] = (__bf16)u.stage[k * DIN + n];
    }
    for (int idx = tid; idx < DIN * 3; idx += 256) sW2[idx] = h_w2[idx];
    if (tid < DIN) sB1[tid] = h_b1[tid];
    if (tid < 3)   sB2[tid] = h_b2[tid];
    __syncthreads();

    int wave = tid >> 5, lane = tid & 31;
    int half = lane >> 4, mrow = lane & 15;
    int row0 = (blockIdx.x * 8 + wave) * 16;
    int r = imin(row0 + mrow, N_NODES - 1);            // clamp, guard stores
    const float* xrow = x + r * DIN;

    v8f z = {};
    v8f acc[8];
#pragma unroll
    for (int nt = 0; nt < 8; ++nt) acc[nt] = z;

#pragma unroll
    for (int kb = 0; kb < 4; ++kb) {
        v16bf a = load_a_frag_f32(xrow, kb, half);
#pragma unroll
        for (int nt = 0; nt < 8; ++nt) {
            v16bf b = load_b_frag(sW1t, DIN, nt, kb, lane);
            acc[nt] = wmma_bf16(a, b, acc[nt]);
        }
    }
    __syncthreads();   // staging buffer is dead; reuse as H

    // hidden = relu(acc + b1) -> LDS (bf16), per-wave tile
    __bf16* hme = u.H[wave];
    int n = lane & 15;
#pragma unroll
    for (int nt = 0; nt < 8; ++nt) {
#pragma unroll
        for (int rr = 0; rr < 8; ++rr) {
            int m = rr + 8 * half;
            float v = acc[nt][rr] + sB1[nt * 16 + n];
            hme[m * DIN + nt * 16 + n] = (__bf16)(v > 0.f ? v : 0.f);
        }
    }
    __syncthreads();

    // delta = tanh(hidden @ h_w2 + b2) : 16 rows x 3 cols, tiny -> VALU
    for (int o = lane; o < 48; o += 32) {
        int rr = o / 3, c = o - rr * 3;
        float s = sB2[c];
        for (int k = 0; k < DIN; ++k)
            s += (float)hme[rr * DIN + k] * sW2[k * 3 + c];
        int gr = row0 + rr;
        if (gr < N_NODES) delta[gr * 3 + c] = tanhf(s);
    }
}

// relative-position feature fragment (K = 128..159, only first 3 real)
__device__ __forceinline__ v16bf geo_frag(const float* __restrict__ pos,
                                          const float* __restrict__ delta,
                                          int s, int d, int half) {
    v16bf a;
#pragma unroll
    for (int i = 0; i < 16; ++i) {
        int kk = a_koff(i, half);
        float v = 0.f;
        if (kk < 3) v = pos[s * 3 + kk] - pos[d * 3 + kk] + delta[d * 3 + kk];
        a[i] = (__bf16)v;
    }
    return a;
}

// relu(acc + bias) scatter-added into agg[dst] for one 16-edge tile
__device__ __forceinline__ void scatter_tile(const v8f acc[8], int e0t, int half,
                                             int n, const long long* __restrict__ ei,
                                             const float* sFB, float* __restrict__ agg) {
#pragma unroll
    for (int rr = 0; rr < 8; ++rr) {
        int ee = e0t + rr + 8 * half;   // edge owning output row rr+8*half
        if (ee < N_EDGES) {
            int dd = (int)ei[N_EDGES + ee];
            float* arow = agg + dd * DOUT;
#pragma unroll
            for (int nt = 0; nt < 8; ++nt) {
                float v = acc[nt][rr] + sFB[nt * 16 + n];
                if (v > 0.f) atomicAdd(arow + nt * 16 + n, v);  // adding 0 is a no-op
            }
        }
    }
}

// ---------------- fused edge: gather -> GEMM(131x128) -> relu -> scatter-add
// 2 M-tiles (32 edges) per wave: every B fragment feeds two WMMAs.
__global__ void __launch_bounds__(256) k_edge(
    const float* __restrict__ x,   const float* __restrict__ pos,
    const long long* __restrict__ ei,
    const float* __restrict__ f_w, const float* __restrict__ f_b,
    const float* __restrict__ delta, float* __restrict__ agg)
{
    const int KP = 160;                                 // 131 padded to 5 K-chunks
    __shared__ __align__(32) __bf16 sFWt[DOUT * KP];    // transposed [n][k], 40KB
    __shared__ float sFB[DOUT];
    __shared__ __align__(16) float sStage[131 * DOUT];  // 65.5KB raw f_w staging

    int tid = threadIdx.x;
    for (int c = tid; c < (131 * DOUT) / 4; c += 256)
        cp_b128(f_w + c * 4, sStage + c * 4);
    cp_wait();
    __syncthreads();
    for (int idx = tid; idx < DOUT * KP; idx += 256) {
        int n = idx / KP, k = idx - n * KP;
        sFWt[idx] = (k < 131) ? (__bf16)sStage[k * DOUT + n] : (__bf16)0.f;
    }
    if (tid < DOUT) sFB[tid] = f_b[tid];
    __syncthreads();

    int wave = tid >> 5, lane = tid & 31;
    int half = lane >> 4, mrow = lane & 15;
    int e0 = (blockIdx.x * 8 + wave) * 32;              // 32 edges per wave
    int eA = imin(e0 + mrow,      N_EDGES - 1);
    int eB = imin(e0 + 16 + mrow, N_EDGES - 1);
    int s0 = (int)ei[eA], d0 = (int)ei[N_EDGES + eA];
    int s1 = (int)ei[eB], d1 = (int)ei[N_EDGES + eB];
    const float* xs0 = x + s0 * DIN;
    const float* xs1 = x + s1 * DIN;

    v8f z = {};
    v8f acc0[8], acc1[8];
#pragma unroll
    for (int nt = 0; nt < 8; ++nt) { acc0[nt] = z; acc1[nt] = z; }

    // K = 0..127 : x[src]
#pragma unroll
    for (int kb = 0; kb < 4; ++kb) {
        v16bf a0 = load_a_frag_f32(xs0, kb, half);
        v16bf a1 = load_a_frag_f32(xs1, kb, half);
#pragma unroll
        for (int nt = 0; nt < 8; ++nt) {
            v16bf b = load_b_frag(sFWt, KP, nt, kb, lane);
            acc0[nt] = wmma_bf16(a0, b, acc0[nt]);
            acc1[nt] = wmma_bf16(a1, b, acc1[nt]);
        }
    }
    // K = 128..159 : [pos_j - pos_i + delta_i, 0...]
    {
        v16bf a0 = geo_frag(pos, delta, s0, d0, half);
        v16bf a1 = geo_frag(pos, delta, s1, d1, half);
#pragma unroll
        for (int nt = 0; nt < 8; ++nt) {
            v16bf b = load_b_frag(sFWt, KP, nt, 4, lane);
            acc0[nt] = wmma_bf16(a0, b, acc0[nt]);
            acc1[nt] = wmma_bf16(a1, b, acc1[nt]);
        }
    }

    int n = lane & 15;
    scatter_tile(acc0, e0,      half, n, ei, sFB, agg);
    scatter_tile(acc1, e0 + 16, half, n, ei, sFB, agg);
}

// ---------------------------------- out = x + relu(relu(agg@g_w1)@g_w2)
__global__ void __launch_bounds__(128) k_node_g(
    const float* __restrict__ x,   const float* __restrict__ agg,
    const float* __restrict__ g_w1, const float* __restrict__ g_b1,
    const float* __restrict__ g_w2, const float* __restrict__ g_b2,
    float* __restrict__ out)
{
    __shared__ __align__(32) __bf16 sG1t[DOUT * DOUT];  // 32KB
    __shared__ __align__(32) __bf16 sG2t[DOUT * DIN];   // 32KB
    __shared__ float sB1[DOUT];
    __shared__ float sB2[DIN];
    __shared__ __align__(32) union {
        float  stage[DOUT * DOUT];   // 64KB raw weight staging (async copy)
        __bf16 T[4][16 * DOUT];      // per-wave mid tile (after staging done)
    } u;

    int tid = threadIdx.x;
    // stage + transpose g_w1
    for (int c = tid; c < (DOUT * DOUT) / 4; c += 128)
        cp_b128(g_w1 + c * 4, u.stage + c * 4);
    cp_wait();
    __syncthreads();
    for (int idx = tid; idx < DOUT * DOUT; idx += 128) {
        int n = idx & 127, k = idx >> 7;
        sG1t[n * DOUT + k] = (__bf16)u.stage[k * DOUT + n];
    }
    __syncthreads();
    // stage + transpose g_w2 (reuse the same staging buffer)
    for (int c = tid; c < (DOUT * DIN) / 4; c += 128)
        cp_b128(g_w2 + c * 4, u.stage + c * 4);
    cp_wait();
    __syncthreads();
    for (int idx = tid; idx < DOUT * DIN; idx += 128) {
        int n = idx & 127, k = idx >> 7;
        sG2t[n * DOUT + k] = (__bf16)u.stage[k * DIN + n];
    }
    if (tid < DOUT) { sB1[tid] = g_b1[tid]; sB2[tid] = g_b2[tid]; }
    __syncthreads();

    int wave = tid >> 5, lane = tid & 31;
    int half = lane >> 4, mrow = lane & 15;
    int row0 = (blockIdx.x * 4 + wave) * 16;
    int r = imin(row0 + mrow, N_NODES - 1);
    const float* arow = agg + r * DOUT;

    v8f z = {};
    v8f acc[8];
#pragma unroll
    for (int nt = 0; nt < 8; ++nt) acc[nt] = z;

    // t = relu(agg @ g_w1 + b1)
#pragma unroll
    for (int kb = 0; kb < 4; ++kb) {
        v16bf a = load_a_frag_f32(arow, kb, half);
#pragma unroll
        for (int nt = 0; nt < 8; ++nt) {
            v16bf b = load_b_frag(sG1t, DOUT, nt, kb, lane);
            acc[nt] = wmma_bf16(a, b, acc[nt]);
        }
    }
    __syncthreads();   // staging buffer dead; reuse as T
    __bf16* tme = u.T[wave];
    int n = lane & 15;
#pragma unroll
    for (int nt = 0; nt < 8; ++nt) {
#pragma unroll
        for (int rr = 0; rr < 8; ++rr) {
            int m = rr + 8 * half;
            float v = acc[nt][rr] + sB1[nt * 16 + n];
            tme[m * DOUT + nt * 16 + n] = (__bf16)(v > 0.f ? v : 0.f);
        }
    }
    __syncthreads();

    // out = x + relu(t @ g_w2 + b2)
#pragma unroll
    for (int nt = 0; nt < 8; ++nt) acc[nt] = z;
#pragma unroll
    for (int kb = 0; kb < 4; ++kb) {
        v16bf a = load_a_frag_lds(tme + mrow * DOUT, kb, half);
#pragma unroll
        for (int nt = 0; nt < 8; ++nt) {
            v16bf b = load_b_frag(sG2t, DOUT, nt, kb, lane);
            acc[nt] = wmma_bf16(a, b, acc[nt]);
        }
    }
#pragma unroll
    for (int nt = 0; nt < 8; ++nt) {
#pragma unroll
        for (int rr = 0; rr < 8; ++rr) {
            int gr = row0 + rr + 8 * half;
            if (gr < N_NODES) {
                int c = nt * 16 + n;
                float v = acc[nt][rr] + sB2[c];
                v = v > 0.f ? v : 0.f;
                out[gr * DIN + c] = x[gr * DIN + c] + v;
            }
        }
    }
}

extern "C" void kernel_launch(void* const* d_in, const int* in_sizes, int n_in,
                              void* d_out, int out_size, void* d_ws, size_t ws_size,
                              hipStream_t stream)
{
    const float*     x    = (const float*)d_in[0];
    const float*     pos  = (const float*)d_in[1];
    const long long* ei   = (const long long*)d_in[2];   // int64 [2, E]
    const float*     h_w1 = (const float*)d_in[3];
    const float*     h_b1 = (const float*)d_in[4];
    const float*     h_w2 = (const float*)d_in[5];
    const float*     h_b2 = (const float*)d_in[6];
    const float*     f_w  = (const float*)d_in[7];
    const float*     f_b  = (const float*)d_in[8];
    const float*     g_w1 = (const float*)d_in[9];
    const float*     g_b1 = (const float*)d_in[10];
    const float*     g_w2 = (const float*)d_in[11];
    const float*     g_b2 = (const float*)d_in[12];

    float* agg   = (float*)d_ws;                         // [N, DOUT] 25.6 MB
    float* delta = agg + (size_t)N_NODES * DOUT;         // [N, 3]
    float* outp  = (float*)d_out;

    int zn = N_NODES * DOUT;
    k_zero<<<(zn + 255) / 256, 256, 0, stream>>>(agg, zn);

    int hBlocks = (N_NODES + 127) / 128;                 // 8 waves x 16 rows
    k_node_h<<<hBlocks, 256, 0, stream>>>(x, h_w1, h_b1, h_w2, h_b2, delta);

    int eBlocks = (N_EDGES + 255) / 256;                 // 8 waves x 32 edges
    k_edge<<<eBlocks, 256, 0, stream>>>(x, pos, ei, f_w, f_b, delta, agg);

    int gBlocks = (N_NODES + 63) / 64;                   // 4 waves x 16 rows
    k_node_g<<<gBlocks, 128, 0, stream>>>(x, agg, g_w1, g_b1, g_w2, g_b2, outp);
}
